// GatherModel_2035814498956
// MI455X (gfx1250) — matrix operand order repaired and checked by reading, not compiled, not run on hardware
//
#include <hip/hip_runtime.h>

#define NN 50000      // nodes
#define NE 100000     // edges
#define DN 40         // node dim
#define DE 10         // edge dim
#define DH 128        // edge hidden dim
#define NSTEPS 6
#define DD 1600       // DN*DN

typedef __attribute__((ext_vector_type(16))) _Float16 v16h;
typedef __attribute__((ext_vector_type(8)))  _Float16 v8h;
typedef __attribute__((ext_vector_type(8)))  float    v8f;

union V16H { v16h v; v8h h[2]; };

// ---------------- zero scratch ----------------
__global__ void k_zero(float* __restrict__ p, int n) {
    int i = blockIdx.x * blockDim.x + threadIdx.x;
    if (i < n) p[i] = 0.f;
}

// ---------------- edge hidden: H = relu(e_feat @ e1_w + e1_b), f16 ----------------
__global__ void k_edge_hidden(const float* __restrict__ e_feat,
                              const float* __restrict__ e1_w,
                              const float* __restrict__ e1_b,
                              _Float16* __restrict__ H) {
    int i = blockIdx.x * blockDim.x + threadIdx.x;   // NE*DH threads
    if (i >= NE * DH) return;
    int e = i >> 7, k = i & 127;
    float acc = e1_b[k];
    #pragma unroll
    for (int j = 0; j < DE; ++j) acc += e_feat[e * DE + j] * e1_w[j * DH + k];
    H[i] = (_Float16)(acc > 0.f ? acc : 0.f);
}

// ---------------- pack e2_w into WMMA B-panel layout (f16) ----------------
// Bp[((ntile*4 + kt)*32 + lane)*16 + j] = e2_w[(kt*32 + (lane<16?0:16) + j)*1600 + ntile*16 + (lane&15)]
__global__ void k_pack_b(const float* __restrict__ e2_w, _Float16* __restrict__ Bp) {
    int i = blockIdx.x * blockDim.x + threadIdx.x;   // 100*4*32*16 = 204800
    if (i >= 100 * 4 * 32 * 16) return;
    int j     =  i        & 15;
    int lane  = (i >> 4)  & 31;
    int kt    = (i >> 9)  & 3;
    int ntile =  i >> 11;
    int n = ntile * 16 + (lane & 15);
    int K = kt * 32 + ((lane < 16) ? 0 : 16) + j;
    Bp[i] = (_Float16)e2_w[K * DD + n];
}

// ---------------- big GEMM: W = H(E x 128) @ e2_w(128 x 1600) + e2_b, via WMMA ----------------
// one wave per 16x16 output tile; 6250 x 100 tiles, K-loop of 4 x v_wmma_f32_16x16x32_f16
__global__ void k_wmma_edge_net(const _Float16* __restrict__ H,
                                const _Float16* __restrict__ Bp,
                                const float* __restrict__ e2_b,
                                _Float16* __restrict__ W) {
    int lane = threadIdx.x & 31;
    int wid  = (blockIdx.x * blockDim.x + threadIdx.x) >> 5;   // exactly 625000 waves
    int etile = wid / 100;
    int ntile = wid % 100;

    int col = ntile * 16 + (lane & 15);
    float bias = e2_b[col];
    v8f c;
    #pragma unroll
    for (int r = 0; r < 8; ++r) c[r] = bias;

    // A operand: row M = lane&15; lanes<16 carry K {0..15}, lanes>=16 carry K {16..31}
    const _Float16* hrow = H + (size_t)(etile * 16 + (lane & 15)) * DH;
    #pragma unroll
    for (int kt = 0; kt < 4; ++kt) {
        int kbase = kt * 32 + ((lane < 16) ? 0 : 16);
        V16H a, b;
        a.h[0] = *(const v8h*)(hrow + kbase);        // 16B chunk, K = kbase..kbase+7
        a.h[1] = *(const v8h*)(hrow + kbase + 8);    // 16B chunk, K = kbase+8..+15
        const _Float16* bp = Bp + ((size_t)(ntile * 4 + kt) * 32 + lane) * 16;
        b.h[0] = *(const v8h*)(bp);                  // 16 contiguous K-values, 32B
        b.h[1] = *(const v8h*)(bp + 8);
        c = __builtin_amdgcn_wmma_f32_16x16x32_f16(false, a.v, false, b.v,
                                                   (short)0, c, false, false);
    }

    // D layout: VGPR r, lanes 0-15 -> row r, lanes 16-31 -> row r+8; col = lane&15
    int rbase = etile * 16 + ((lane < 16) ? 0 : 8);
    #pragma unroll
    for (int r = 0; r < 8; ++r)
        W[(size_t)(rbase + r) * DD + col] = (_Float16)c[r];
}

// ---------------- lin0: out = relu(n_feat @ lin0_w + lin0_b) ----------------
__global__ void k_lin0(const float* __restrict__ n_feat, const float* __restrict__ w,
                       const float* __restrict__ b, float* __restrict__ out) {
    __shared__ float xs[8][DN];
    int tx = threadIdx.x;                 // 320 threads = 8 nodes x 40
    int ln = tx / DN, o = tx % DN;
    int n = blockIdx.x * 8 + ln;          // NN divisible by 8
    xs[ln][o] = n_feat[n * DN + o];
    __syncthreads();
    float acc = b[o];
    #pragma unroll
    for (int d = 0; d < DN; ++d) acc += xs[ln][d] * w[d * DN + o];
    out[n * DN + o] = acc > 0.f ? acc : 0.f;
}

// ---------------- per-edge matvec + scatter: agg[dst] += x[src] @ W_e ----------------
__global__ void k_matvec_scatter(const float* __restrict__ x, const _Float16* __restrict__ W,
                                 const int* __restrict__ src, const int* __restrict__ dst,
                                 float* __restrict__ agg) {
    __shared__ float xs[8][DN];
    int tx = threadIdx.x;                 // 320 threads = 8 edges x 40
    int le = tx / DN, o = tx % DN;
    int e = blockIdx.x * 8 + le;          // NE divisible by 8
    xs[le][o] = x[src[e] * DN + o];
    __syncthreads();
    const _Float16* wrow = W + (size_t)e * DD + o;   // layout (e, d, o): coalesced in o
    float acc = 0.f;
    #pragma unroll 8
    for (int d = 0; d < DN; ++d) acc += xs[le][d] * (float)wrow[d * DN];
    atomicAdd(&agg[dst[e] * DN + o], acc);
}

// ---------------- node update: m = relu(agg + x@res_w + conv_b); xnew = [m,x]@msg_w + msg_b ----------------
__global__ void k_node_update(const float* __restrict__ x, float* __restrict__ agg,
                              const float* __restrict__ res_w, const float* __restrict__ conv_b,
                              const float* __restrict__ msg_w, const float* __restrict__ msg_b,
                              float* __restrict__ xnew) {
    __shared__ float xs[8][DN];
    __shared__ float ms[8][DN];
    int tx = threadIdx.x;                 // 320 threads = 8 nodes x 40
    int ln = tx / DN, o = tx % DN;
    int n = blockIdx.x * 8 + ln;
    xs[ln][o] = x[n * DN + o];
    __syncthreads();
    {
        float acc = conv_b[o] + agg[n * DN + o];
        agg[n * DN + o] = 0.f;            // self-clean for next step
        #pragma unroll
        for (int d = 0; d < DN; ++d) acc += xs[ln][d] * res_w[d * DN + o];
        ms[ln][o] = acc > 0.f ? acc : 0.f;
    }
    __syncthreads();
    float acc = msg_b[o];
    #pragma unroll
    for (int d = 0; d < DN; ++d) acc += ms[ln][d] * msg_w[d * DN + o];
    #pragma unroll
    for (int d = 0; d < DN; ++d) acc += xs[ln][d] * msg_w[(DN + d) * DN + o];
    xnew[n * DN + o] = acc;
}

// ---------------- epilogue: out = x + n_feat ----------------
__global__ void k_epilogue(const float* __restrict__ x, const float* __restrict__ n_feat,
                           float* __restrict__ out, int n) {
    int i = blockIdx.x * blockDim.x + threadIdx.x;
    if (i < n) out[i] = x[i] + n_feat[i];
}

extern "C" void kernel_launch(void* const* d_in, const int* in_sizes, int n_in,
                              void* d_out, int out_size, void* d_ws, size_t ws_size,
                              hipStream_t stream) {
    (void)in_sizes; (void)n_in; (void)out_size; (void)ws_size;
    const float* n_feat = (const float*)d_in[0];
    const float* e_feat = (const float*)d_in[1];
    const int*   src    = (const int*)  d_in[2];
    const int*   dst    = (const int*)  d_in[3];
    const float* lin0_w = (const float*)d_in[4];
    const float* lin0_b = (const float*)d_in[5];
    const float* msg_w  = (const float*)d_in[6];
    const float* msg_b  = (const float*)d_in[7];
    const float* e1_w   = (const float*)d_in[8];
    const float* e1_b   = (const float*)d_in[9];
    const float* e2_w   = (const float*)d_in[10];
    const float* e2_b   = (const float*)d_in[11];
    const float* res_w  = (const float*)d_in[12];
    const float* conv_b = (const float*)d_in[13];

    char* ws = (char*)d_ws;
    size_t off = 0;
    auto alloc = [&](size_t bytes) -> void* {
        void* p = ws + off;
        off = (off + bytes + 255) & ~(size_t)255;
        return p;
    };
    _Float16* W   = (_Float16*)alloc((size_t)NE * DD * sizeof(_Float16));   // 320 MB
    _Float16* H   = (_Float16*)alloc((size_t)NE * DH * sizeof(_Float16));   // 25.6 MB
    _Float16* Bp  = (_Float16*)alloc((size_t)204800 * sizeof(_Float16));    // 0.4 MB
    float*    agg = (float*)alloc((size_t)NN * DN * sizeof(float));         // 8 MB
    float*    oA  = (float*)alloc((size_t)NN * DN * sizeof(float));         // 8 MB
    float*    oB  = (float*)alloc((size_t)NN * DN * sizeof(float));         // 8 MB

    // precompute (edge network, staged for WMMA)
    k_zero<<<(NN * DN + 255) / 256, 256, 0, stream>>>(agg, NN * DN);
    k_edge_hidden<<<(NE * DH + 255) / 256, 256, 0, stream>>>(e_feat, e1_w, e1_b, H);
    k_pack_b<<<(204800 + 255) / 256, 256, 0, stream>>>(e2_w, Bp);
    k_wmma_edge_net<<<(6250 * 100) / 8, 256, 0, stream>>>(H, Bp, e2_b, W);   // 8 waves/block

    // node init
    k_lin0<<<NN / 8, 320, 0, stream>>>(n_feat, lin0_w, lin0_b, oA);

    // message-passing loop
    float* cur = oA; float* nxt = oB;
    for (int s = 0; s < NSTEPS; ++s) {
        k_matvec_scatter<<<NE / 8, 320, 0, stream>>>(cur, W, src, dst, agg);
        k_node_update<<<NN / 8, 320, 0, stream>>>(cur, agg, res_w, conv_b, msg_w, msg_b, nxt);
        float* t = cur; cur = nxt; nxt = t;
    }
    k_epilogue<<<(NN * DN + 255) / 256, 256, 0, stream>>>(cur, n_feat, (float*)d_out, NN * DN);
}
